// RNN_69844758168129
// MI455X (gfx1250) — matrix operand order repaired and checked by reading, not compiled, not run on hardware
//
#include <hip/hip_runtime.h>
#include <math.h>

// Problem constants (match reference)
#define VSZ 10000
#define ESZ 512
#define HSZ 512
#define LSZ 2
#define SSZ 128
#define BSZ 128

typedef __attribute__((ext_vector_type(16))) __bf16       v16bf;
typedef __attribute__((ext_vector_type(8)))  float        v8f;
typedef __attribute__((ext_vector_type(4)))  unsigned int v4u;

union FragU { v16bf v; v4u q[2]; };

__device__ __forceinline__ unsigned short f32_to_bf16(float x) {
    unsigned int u = __float_as_uint(x);
    u += 0x7FFFu + ((u >> 16) & 1u);   // round-to-nearest-even
    return (unsigned short)(u >> 16);
}
__device__ __forceinline__ float bf16_to_f32(unsigned short h) {
    return __uint_as_float(((unsigned int)h) << 16);
}

// ---- WMMA fragment loads from GLOBAL (bf16, 16x16x32) -------------------
// A (16x32 MxK): lane<16: K chunks [k0..k0+7],[k0+16..k0+23]; lane>=16: +8.
__device__ __forceinline__ v16bf load_fragA(const unsigned short* rowbase, int k0, int lane) {
    int kk = k0 + ((lane >> 4) & 1) * 8;
    const unsigned short* p = rowbase + kk;
    FragU f;
    f.q[0] = *(const v4u*)(p);
    f.q[1] = *(const v4u*)(p + 16);
    return f.v;
}
// B (32x16 KxN), B[k][n] = W[n][k], W row-major [N,K] stride ldw.
// lane holds column n=lane&15; K chunks [k0..k0+7],[k0+16..k0+23] (lane>=16 -> +8).
__device__ __forceinline__ v16bf load_fragB(const unsigned short* W, int ldw,
                                            int n0, int k0, int lane) {
    int kk = k0 + ((lane >> 4) & 1) * 8;
    const unsigned short* p = W + (size_t)(n0 + (lane & 15)) * ldw + kk;
    FragU f;
    f.q[0] = *(const v4u*)(p);
    f.q[1] = *(const v4u*)(p + 16);
    return f.v;
}

#define WMMA_BF16(a, b, c) \
    __builtin_amdgcn_wmma_f32_16x16x32_bf16(false, (a), false, (b), (short)0, (c), false, false)

// CDNA5 async global->LDS copy (16B per lane, tracked by ASYNCcnt)
__device__ __forceinline__ void async_copy_b128(unsigned lds_off, const unsigned short* g) {
    asm volatile("global_load_async_to_lds_b128 %0, %1, off"
                 :: "v"(lds_off), "v"((unsigned long long)(uintptr_t)g)
                 : "memory");
}
__device__ __forceinline__ void wait_async_all() {
    asm volatile("s_wait_asynccnt 0x0" ::: "memory");
}

// ---- elementwise converters ---------------------------------------------
__global__ void cvt_f32_to_bf16(const float* __restrict__ src,
                                unsigned short* __restrict__ dst, int n) {
    int i = blockIdx.x * blockDim.x + threadIdx.x;
    if (i < n) dst[i] = f32_to_bf16(src[i]);
}
__global__ void cvt_bf16_to_f32(const unsigned short* __restrict__ src,
                                float* __restrict__ dst, int n) {
    int i = blockIdx.x * blockDim.x + threadIdx.x;
    if (i < n) dst[i] = bf16_to_f32(src[i]);
}

// ---- one RNN layer for one timestep -------------------------------------
// hNew[b,j] = tanh( A[b,:] @ Wih[j,:] + hOld[b,:] @ Whh[j,:] + bias[j] )
// Grid: <<<32, 256>>>; all 8 waves of a WG share one N-tile (weight reuse in
// WGP$), waves differ in M.
__global__ __launch_bounds__(256) void rnn_layer_step(
        const unsigned short* __restrict__ actIn,
        const int* __restrict__ tok,
        const unsigned short* __restrict__ embBf,
        const unsigned short* __restrict__ hOld,
        const unsigned short* __restrict__ Wih,
        const unsigned short* __restrict__ Whh,
        const float* __restrict__ bias,
        unsigned short* __restrict__ hNew) {
    int lane = threadIdx.x & 31;
    int mt   = threadIdx.x >> 5;      // 0..7  (M tile)
    int nt   = blockIdx.x;            // 0..31 (N tile, shared by the WG)
    int m0 = mt * 16, n0 = nt * 16;

    int m = m0 + (lane & 15);
    const unsigned short* arow =
        tok ? (embBf + (size_t)tok[m] * HSZ) : (actIn + (size_t)m * HSZ);
    const unsigned short* hrow = hOld + (size_t)m * HSZ;

    v8f acc = {0.f, 0.f, 0.f, 0.f, 0.f, 0.f, 0.f, 0.f};
#pragma unroll 4
    for (int k0 = 0; k0 < HSZ; k0 += 32) {
        v16bf af = load_fragA(arow, k0, lane);
        v16bf bi = load_fragB(Wih, HSZ, n0, k0, lane);
        acc = WMMA_BF16(af, bi, acc);
        v16bf hf = load_fragA(hrow, k0, lane);
        v16bf bh = load_fragB(Whh, HSZ, n0, k0, lane);
        acc = WMMA_BF16(hf, bh, acc);
    }
    int n  = n0 + (lane & 15);
    float bb = bias[n];
    int mb = m0 + 8 * ((lane >> 4) & 1);
#pragma unroll
    for (int r = 0; r < 8; r++) {
        float v = tanhf(acc[r] + bb);
        hNew[(size_t)(mb + r) * HSZ + n] = f32_to_bf16(v);
    }
}

// ---- big output projection ----------------------------------------------
// logits[M=S*B=16384, N=V=10000] = hAll[M,512] @ Wout[N,512]^T + bOut
// WG tile 256M x 80N, K-slab 32, double-buffered LDS via async copies.
// 8 waves: wave w owns rows [w*32, w*32+32), all 5 N tiles (B frags shared).
#define MBLK 256
#define NBLK 80
#define KSLAB 32
#define PADW 40          // row stride in elements (80B = 20 banks, conflict-free)
#define NCHUNK ((MBLK + NBLK) * KSLAB / 8)   // 16B chunks per slab = 1344

__global__ __launch_bounds__(256) void out_proj_gemm(
        const unsigned short* __restrict__ hAll,
        const unsigned short* __restrict__ Wout,
        const float* __restrict__ bOut,
        float* __restrict__ logits) {
    alignas(16) __shared__ unsigned short ldsA[2][MBLK][PADW];
    alignas(16) __shared__ unsigned short ldsB[2][NBLK][PADW];

    int tid  = threadIdx.x;
    int lane = tid & 31;
    int wv   = tid >> 5;                       // 0..7
    int mb   = blockIdx.x / (VSZ / NBLK);      // 64 M blocks
    int nbk  = blockIdx.x % (VSZ / NBLK);      // 125 N blocks
    int m0 = mb * MBLK, n0 = nbk * NBLK;

    auto issue_slab = [&](int buf, int k0) {
        for (int idx = tid; idx < NCHUNK; idx += 256) {
            unsigned loff;
            const unsigned short* g;
            if (idx < MBLK * 4) {              // A: 4 chunks per 32-elem row
                int row = idx >> 2, c = idx & 3;
                g    = hAll + (size_t)(m0 + row) * HSZ + k0 + c * 8;
                loff = (unsigned)(uintptr_t)&ldsA[buf][row][c * 8];
            } else {                           // B
                int j = idx - MBLK * 4;
                int row = j >> 2, c = j & 3;
                g    = Wout + (size_t)(n0 + row) * HSZ + k0 + c * 8;
                loff = (unsigned)(uintptr_t)&ldsB[buf][row][c * 8];
            }
            async_copy_b128(loff, g);
        }
    };

    v8f zero8 = {0.f, 0.f, 0.f, 0.f, 0.f, 0.f, 0.f, 0.f};
    v8f acc[2][5];
#pragma unroll
    for (int s = 0; s < 2; s++)
#pragma unroll
        for (int j = 0; j < 5; j++) acc[s][j] = zero8;

    issue_slab(0, 0);

    for (int ks = 0; ks < HSZ / KSLAB; ++ks) {   // 16 K-slabs
        int cur = ks & 1;
        wait_async_all();          // this wave's copies into `cur` are done
        __syncthreads();           // everyone's copies visible
        if (ks + 1 < HSZ / KSLAB) issue_slab(cur ^ 1, (ks + 1) * KSLAB);

        // fragment loads from LDS + 10 WMMAs
        int kk = ((lane >> 4) & 1) * 8;
        int kb = ((lane >> 4) & 1) * 8;
        const unsigned short* ar0 = &ldsA[cur][wv * 32 + (lane & 15)][0];
        const unsigned short* ar1 = &ldsA[cur][wv * 32 + 16 + (lane & 15)][0];
        FragU fa0, fa1;
        fa0.q[0] = *(const v4u*)(ar0 + kk);
        fa0.q[1] = *(const v4u*)(ar0 + kk + 16);
        fa1.q[0] = *(const v4u*)(ar1 + kk);
        fa1.q[1] = *(const v4u*)(ar1 + kk + 16);
        v16bf bf[5];
#pragma unroll
        for (int j = 0; j < 5; j++) {
            const unsigned short* br = &ldsB[cur][j * 16 + (lane & 15)][kb];
            FragU fb;
            fb.q[0] = *(const v4u*)(br);
            fb.q[1] = *(const v4u*)(br + 16);
            bf[j] = fb.v;
        }
#pragma unroll
        for (int j = 0; j < 5; j++) {
            acc[0][j] = WMMA_BF16(fa0.v, bf[j], acc[0][j]);
            acc[1][j] = WMMA_BF16(fa1.v, bf[j], acc[1][j]);
        }
        __syncthreads();           // all reads of `cur` done before it refills
    }

    // epilogue: bias + f32 store
    int rbase = m0 + wv * 32 + 8 * ((lane >> 4) & 1);
#pragma unroll
    for (int j = 0; j < 5; j++) {
        int n = n0 + j * 16 + (lane & 15);
        float bb = bOut[n];
#pragma unroll
        for (int s = 0; s < 2; s++)
#pragma unroll
            for (int r = 0; r < 8; r++)
                logits[(size_t)(rbase + s * 16 + r) * VSZ + n] = acc[s][j][r] + bb;
    }
}

extern "C" void kernel_launch(void* const* d_in, const int* in_sizes, int n_in,
                              void* d_out, int out_size, void* d_ws, size_t ws_size,
                              hipStream_t stream) {
    const int*   tokens = (const int*)  d_in[0];   // [S,B]
    const float* hidden = (const float*)d_in[1];   // [L,B,H]
    const float* emb    = (const float*)d_in[2];   // [V,E]
    const float* W_ih   = (const float*)d_in[3];   // [L,H,H]
    const float* W_hh   = (const float*)d_in[4];   // [L,H,H]
    const float* b_hh   = (const float*)d_in[5];   // [L,H]
    const float* W_out  = (const float*)d_in[6];   // [V,H]
    const float* b_out  = (const float*)d_in[7];   // [V]

    // ---- workspace carving (all chunks 256B-aligned) ----
    char* ws = (char*)d_ws;
    unsigned short* wWih  = (unsigned short*)ws; ws += (size_t)LSZ*HSZ*HSZ*2;  // 1 MB
    unsigned short* wWhh  = (unsigned short*)ws; ws += (size_t)LSZ*HSZ*HSZ*2;  // 1 MB
    unsigned short* wWout = (unsigned short*)ws; ws += (size_t)VSZ*HSZ*2;      // 10 MB
    unsigned short* wEmb  = (unsigned short*)ws; ws += (size_t)VSZ*ESZ*2;      // 10 MB
    unsigned short* hAll  = (unsigned short*)ws; ws += (size_t)SSZ*BSZ*HSZ*2;  // 16 MB
    unsigned short* h0a   = (unsigned short*)ws; ws += (size_t)BSZ*HSZ*2;
    unsigned short* h0b   = (unsigned short*)ws; ws += (size_t)BSZ*HSZ*2;
    unsigned short* hInit1= (unsigned short*)ws; ws += (size_t)BSZ*HSZ*2;
    unsigned short* h0buf[2] = {h0a, h0b};

    // ---- convert weights/embeddings/initial hidden to bf16 ----
    auto cvt = [&](const float* s, unsigned short* d, int n) {
        cvt_f32_to_bf16<<<(n + 255) / 256, 256, 0, stream>>>(s, d, n);
    };
    cvt(W_ih,  wWih,  LSZ*HSZ*HSZ);
    cvt(W_hh,  wWhh,  LSZ*HSZ*HSZ);
    cvt(W_out, wWout, VSZ*HSZ);
    cvt(emb,   wEmb,  VSZ*ESZ);
    cvt(hidden,             h0buf[0], BSZ*HSZ);   // layer-0 initial state
    cvt(hidden + BSZ*HSZ,   hInit1,   BSZ*HSZ);   // layer-1 initial state

    // ---- sequential recurrence: 2 small WMMA GEMM kernels per step ----
    for (int t = 0; t < SSZ; t++) {
        rnn_layer_step<<<32, 256, 0, stream>>>(
            nullptr, tokens + (size_t)t * BSZ, wEmb,
            h0buf[t & 1], wWih, wWhh, b_hh, h0buf[(t + 1) & 1]);
        const unsigned short* h1old = (t == 0) ? hInit1 : (hAll + (size_t)(t - 1) * BSZ * HSZ);
        rnn_layer_step<<<32, 256, 0, stream>>>(
            h0buf[(t + 1) & 1], nullptr, nullptr,
            h1old, wWih + HSZ*HSZ, wWhh + HSZ*HSZ, b_hh + HSZ,
            hAll + (size_t)t * BSZ * HSZ);
    }

    // ---- one big WMMA GEMM for all logits: [16384,512] x [512,10000] ----
    {
        int blocks = (SSZ * BSZ / MBLK) * (VSZ / NBLK);   // 64 * 125 = 8000
        out_proj_gemm<<<blocks, 256, 0, stream>>>(hAll, wWout, b_out, (float*)d_out);
    }

    // ---- final hidden state (f32) appended after logits ----
    float* hfin = (float*)d_out + (size_t)SSZ * BSZ * VSZ;
    cvt_bf16_to_f32<<<(BSZ*HSZ + 255) / 256, 256, 0, stream>>>(h0buf[SSZ & 1], hfin, BSZ*HSZ);
    cvt_bf16_to_f32<<<(BSZ*HSZ + 255) / 256, 256, 0, stream>>>(
        hAll + (size_t)(SSZ - 1) * BSZ * HSZ, hfin + BSZ*HSZ, BSZ*HSZ);
}
